// UniversalTransformers_28123445854587
// MI455X (gfx1250) — compile-verified, matrix operand order checked by reading, not run on hardware
//
#include <hip/hip_runtime.h>
#include <cstdint>
#include <cstddef>

typedef __attribute__((ext_vector_type(16))) _Float16 v16h;
typedef __attribute__((ext_vector_type(8)))  float    v8f;

union FragH { v16h v; uint4 u[2]; };   // 32-byte WMMA half fragment

#define N_NODES 10000
#define N_EDGES 160000
#define DIM     64
#define HEADS   4
#define HD      256            // HEADS*DIM
#define N_PRED  2
#define N_CLASS 16
#define N_STEPS 4
#define E_TOT   (N_EDGES + N_NODES)   // reference appends self-loops

// --- WMMA 16x16x32 f16 fragment k-index pattern (ISA 7.12.2, 16-bit A 16x32) ---
// half index i in 0..15, lane-group g = lane>>4:  k = (i&7) + ((i>>3)<<4) + (g<<3)
__device__ __forceinline__ int frag_k(int i, int g) {
    return (i & 7) + ((i >> 3) << 4) + (g << 3);
}

// ---------------------------------------------------------------------------
// Pre-pack a weight matrix B[K,NC] (f32, row-major) into WMMA B-fragment
// order as f16: for tile tn, k-chunk kk32, lane l, the 16 halves that lane
// feeds to v_wmma are stored contiguously (32 bytes).  Runs once per launch.
// Layout: P[(((tn*k32s)+kk32)*32 + lane)*16 + i]
// ---------------------------------------------------------------------------
__global__ void pack_b(const float* __restrict__ B, _Float16* __restrict__ P,
                       int K, int NC)
{
    int t = blockIdx.x * blockDim.x + threadIdx.x;
    int k32s = K >> 5;
    int total = (NC >> 4) * k32s * 32;
    if (t >= total) return;
    int lane = t & 31;
    int rest = t >> 5;
    int kk32 = rest % k32s;
    int tn   = rest / k32s;
    int g = lane >> 4, ln = lane & 15;
    int n = tn * 16 + ln;
    _Float16* dst = P + (size_t)t * 16;
    #pragma unroll
    for (int i = 0; i < 16; ++i) {
        int k = (kk32 << 5) + frag_k(i, g);
        dst[i] = (_Float16)B[(size_t)k * NC + n];
    }
}

// Build an A fragment from a contiguous f32 row (two 32B runs -> 4x float4).
__device__ __forceinline__ v16h load_a_frag(const float* __restrict__ Arow,
                                            const float* __restrict__ abias,
                                            int kk, int g)
{
    int o0 = kk + (g << 3);
    int o1 = kk + 16 + (g << 3);
    float4 a0 = *(const float4*)(Arow + o0);
    float4 a1 = *(const float4*)(Arow + o0 + 4);
    float4 a2 = *(const float4*)(Arow + o1);
    float4 a3 = *(const float4*)(Arow + o1 + 4);
    if (abias) {
        float4 b0 = *(const float4*)(abias + o0);
        float4 b1 = *(const float4*)(abias + o0 + 4);
        float4 b2 = *(const float4*)(abias + o1);
        float4 b3 = *(const float4*)(abias + o1 + 4);
        a0.x += b0.x; a0.y += b0.y; a0.z += b0.z; a0.w += b0.w;
        a1.x += b1.x; a1.y += b1.y; a1.z += b1.z; a1.w += b1.w;
        a2.x += b2.x; a2.y += b2.y; a2.z += b2.z; a2.w += b2.w;
        a3.x += b3.x; a3.y += b3.y; a3.z += b3.z; a3.w += b3.w;
    }
    v16h af;
    af[0]  = (_Float16)a0.x; af[1]  = (_Float16)a0.y;
    af[2]  = (_Float16)a0.z; af[3]  = (_Float16)a0.w;
    af[4]  = (_Float16)a1.x; af[5]  = (_Float16)a1.y;
    af[6]  = (_Float16)a1.z; af[7]  = (_Float16)a1.w;
    af[8]  = (_Float16)a2.x; af[9]  = (_Float16)a2.y;
    af[10] = (_Float16)a2.z; af[11] = (_Float16)a2.w;
    af[12] = (_Float16)a3.x; af[13] = (_Float16)a3.y;
    af[14] = (_Float16)a3.z; af[15] = (_Float16)a3.w;
    return af;
}

// ---------------------------------------------------------------------------
// Fused GEMM: C[M,NC] = act( (A + abias?)[M,K] @ B[K,NC] + bias )
// One wave32 per 16x(16*NT) output strip: one A fragment is reused across NT
// adjacent column tiles (NT v_wmma per k-chunk).  B comes pre-packed (f16).
// M, NC multiples of 16*NT/16; K multiple of 32.
// ---------------------------------------------------------------------------
template <bool RELU, int NT>
__global__ void gemm_wmma(const float* __restrict__ A, const _Float16* __restrict__ Bp,
                          const float* __restrict__ bias, const float* __restrict__ abias,
                          float* __restrict__ C, int M, int K, int NC)
{
    int tilesN  = NC >> 4;
    int groupsN = tilesN / NT;
    int tilesM  = M >> 4;
    int wave = blockIdx.x * (blockDim.x >> 5) + (threadIdx.x >> 5);
    if (wave >= tilesM * groupsN) return;
    int tm  = wave / groupsN;
    int tn0 = (wave % groupsN) * NT;
    int lane = threadIdx.x & 31;
    int g  = lane >> 4;
    int ln = lane & 15;
    int am = tm * 16 + ln;                       // A row this lane feeds
    int k32s = K >> 5;
    const float* Arow = A + (size_t)am * K;

    v8f acc[NT];
    #pragma unroll
    for (int nt = 0; nt < NT; ++nt)
        #pragma unroll
        for (int r = 0; r < 8; ++r) acc[nt][r] = 0.0f;

    for (int kk32 = 0; kk32 < k32s; ++kk32) {
        v16h af = load_a_frag(Arow, abias, kk32 << 5, g);
        #pragma unroll
        for (int nt = 0; nt < NT; ++nt) {
            const uint4* bp = (const uint4*)(Bp +
                ((((size_t)(tn0 + nt) * k32s + kk32) * 32 + lane) << 4));
            FragH bf; bf.u[0] = bp[0]; bf.u[1] = bp[1];
            acc[nt] = __builtin_amdgcn_wmma_f32_16x16x32_f16(false, af, false, bf.v,
                                                             (short)0, acc[nt], false, false);
        }
    }

    #pragma unroll
    for (int nt = 0; nt < NT; ++nt) {
        int bn = (tn0 + nt) * 16 + ln;           // C column this lane owns
        float cb = bias ? bias[bn] : 0.0f;
        #pragma unroll
        for (int r = 0; r < 8; ++r) {
            int m = tm * 16 + r + g * 8;         // C/D layout: vgpr r -> row r + 8g
            float v = acc[nt][r] + cb;
            if (RELU) v = fmaxf(v, 0.0f);
            C[(size_t)m * NC + bn] = v;
        }
    }
}

// ---------------------------------------------------------------------------
// Fused decoder: logits = final @ dec_W[p] + dec_b[p], then log_softmax over
// the 16 classes (classes live across a 16-lane group).  B pre-packed.
// ---------------------------------------------------------------------------
__global__ void decoder_wmma(const float* __restrict__ F, const _Float16* __restrict__ Pdec,
                             const float* __restrict__ decB, float* __restrict__ out)
{
    int wave   = blockIdx.x * (blockDim.x >> 5) + (threadIdx.x >> 5);
    int tilesM = N_NODES >> 4;                   // 625
    if (wave >= N_PRED * tilesM) return;
    int p  = wave / tilesM;
    int tm = wave % tilesM;
    int lane = threadIdx.x & 31;
    int g  = lane >> 4;
    int ln = lane & 15;                          // class index for this lane
    int am = tm * 16 + ln;
    const float*    Arow = F + (size_t)am * DIM;
    const _Float16* Bp   = Pdec + (size_t)p * DIM * N_CLASS;   // packed, tn==0
    const int k32s = DIM >> 5;                   // 2

    v8f acc;
    #pragma unroll
    for (int r = 0; r < 8; ++r) acc[r] = 0.0f;

    #pragma unroll
    for (int kk32 = 0; kk32 < k32s; ++kk32) {
        v16h af = load_a_frag(Arow, nullptr, kk32 << 5, g);
        const uint4* bp = (const uint4*)(Bp + (((size_t)kk32 * 32 + lane) << 4));
        FragH bf; bf.u[0] = bp[0]; bf.u[1] = bp[1];
        acc = __builtin_amdgcn_wmma_f32_16x16x32_f16(false, af, false, bf.v,
                                                     (short)0, acc, false, false);
    }

    float cb = decB[p * N_CLASS + ln];
    #pragma unroll
    for (int r = 0; r < 8; ++r) {
        float v = acc[r] + cb;
        float mx = v;
        #pragma unroll
        for (int off = 8; off; off >>= 1) mx = fmaxf(mx, __shfl_xor(mx, off, 16));
        float ex = __expf(v - mx);
        float sum = ex;
        #pragma unroll
        for (int off = 8; off; off >>= 1) sum += __shfl_xor(sum, off, 16);
        int m = tm * 16 + r + g * 8;
        out[((size_t)p * N_NODES + m) * N_CLASS + ln] = (v - mx) - __logf(sum);
    }
}

// ---------------------------------------------------------------------------
// Segment-softmax edge kernels (atomics + gathers hit the 192MB L2; the
// xl/xr/agg tables total ~31MB and stay L2-resident)
// ---------------------------------------------------------------------------
__device__ __forceinline__ void atomicMaxFloat(float* addr, float val) {
    int* ia = (int*)addr;
    int old = __float_as_int(*addr);
    while (__int_as_float(old) < val) {
        int assumed = old;
        old = atomicCAS(ia, assumed, __float_as_int(val));
        if (old == assumed) break;
    }
}

__device__ __forceinline__ void edge_ends(const int* __restrict__ ei, int e, int& s, int& d) {
    if (e < N_EDGES) { s = ei[e]; d = ei[N_EDGES + e]; }
    else             { s = d = e - N_EDGES; }           // self-loop
}

// one thread per (edge, head): per-edge GATv2 score + running segment max
__global__ void edge_logits(const int* __restrict__ ei, const float* __restrict__ xl,
                            const float* __restrict__ xr, const float* __restrict__ att,
                            float* __restrict__ logitb, float* __restrict__ mbuf)
{
    __shared__ float s_att[HD];
    for (int i = threadIdx.x; i < HD; i += blockDim.x) s_att[i] = att[i];
    __syncthreads();
    int t = blockIdx.x * blockDim.x + threadIdx.x;
    int e = t >> 2, h = t & 3;
    if (e >= E_TOT) return;
    int s, d; edge_ends(ei, e, s, d);
    const float* pl = xl + (size_t)s * HD + h * DIM;
    const float* pr = xr + (size_t)d * HD + h * DIM;
    const float* pa = s_att + h * DIM;
    float acc = 0.0f;
    #pragma unroll
    for (int c = 0; c < DIM; ++c) {
        float v = pl[c] + pr[c];
        v = (v > 0.0f) ? v : 0.2f * v;                 // leaky_relu(0.2)
        acc += v * pa[c];
    }
    logitb[(size_t)e * HEADS + h] = acc;
    atomicMaxFloat(&mbuf[(size_t)d * HEADS + h], acc);
}

// one thread per (edge, head): exp(logit - segmax) + segment denom
__global__ void edge_exp(const int* __restrict__ ei, float* __restrict__ logitb,
                         const float* __restrict__ mbuf, float* __restrict__ denom)
{
    int t = blockIdx.x * blockDim.x + threadIdx.x;
    int e = t >> 2, h = t & 3;
    if (e >= E_TOT) return;
    int s, d; edge_ends(ei, e, s, d); (void)s;
    float ex = __expf(logitb[(size_t)e * HEADS + h] - mbuf[(size_t)d * HEADS + h]);
    logitb[(size_t)e * HEADS + h] = ex;
    atomicAdd(&denom[(size_t)d * HEADS + h], ex);
}

// one 32-lane group per edge: agg[dst] += alpha * xl[src]  (256 channels)
__global__ void edge_agg(const int* __restrict__ ei, const float* __restrict__ xl,
                         const float* __restrict__ exb, const float* __restrict__ denom,
                         float* __restrict__ agg)
{
    int gt = blockIdx.x * blockDim.x + threadIdx.x;
    int e = gt >> 5, l = gt & 31;
    if (e >= E_TOT) return;
    int s, d; edge_ends(ei, e, s, d);
    float alpha[HEADS];
    #pragma unroll
    for (int h = 0; h < HEADS; ++h)
        alpha[h] = exb[(size_t)e * HEADS + h] / (denom[(size_t)d * HEADS + h] + 1e-16f);
    #pragma unroll
    for (int j = 0; j < 8; ++j) {
        int f = (j << 5) | l;                          // contiguous per lane-step
        atomicAdd(&agg[(size_t)d * HD + f], xl[(size_t)s * HD + f] * alpha[f >> 6]);
    }
}

// ---------------------------------------------------------------------------
// ACT halting update: term = sigmoid(h @ act_W + act_b); final += delta*h
// ---------------------------------------------------------------------------
__global__ void act_update(const float* __restrict__ h, const float* __restrict__ actW,
                           const float* __restrict__ actB, float* __restrict__ total,
                           float* __restrict__ fin)
{
    int gt = blockIdx.x * blockDim.x + threadIdx.x;
    int n = gt >> 5, l = gt & 31;
    if (n >= N_NODES) return;
    float h0 = h[(size_t)n * DIM + l];
    float h1 = h[(size_t)n * DIM + 32 + l];
    float part = h0 * actW[l] + h1 * actW[32 + l];
    #pragma unroll
    for (int off = 16; off; off >>= 1) part += __shfl_xor(part, off, 32);
    float term = 1.0f / (1.0f + __expf(-(part + actB[0])));   // identical in all lanes
    float tot = total[n];
    float nt = fminf(tot + term, 1.0f);
    float delta = fminf(term, nt - tot);
    if (l == 0) total[n] = tot + delta;
    fin[(size_t)n * DIM + l]      += delta * h0;
    fin[(size_t)n * DIM + 32 + l] += delta * h1;
}

// ---------------------------------------------------------------------------
// init kernels (d_ws is poisoned 0xAA; everything read must be initialized)
// ---------------------------------------------------------------------------
__global__ void init_seg(float* __restrict__ agg, float* __restrict__ mbuf,
                         float* __restrict__ denom)
{
    int t = blockIdx.x * blockDim.x + threadIdx.x;
    if (t < N_NODES * HD) agg[t] = 0.0f;
    if (t < N_NODES * HEADS) { mbuf[t] = -3.0e38f; denom[t] = 0.0f; }
}

__global__ void init_state(float* __restrict__ total, float* __restrict__ fin)
{
    int t = blockIdx.x * blockDim.x + threadIdx.x;
    if (t < N_NODES * DIM) fin[t] = 0.0f;
    if (t < N_NODES) total[t] = 0.0f;
}

// ---------------------------------------------------------------------------
extern "C" void kernel_launch(void* const* d_in, const int* in_sizes, int n_in,
                              void* d_out, int out_size, void* d_ws, size_t ws_size,
                              hipStream_t stream)
{
    (void)in_sizes; (void)n_in; (void)out_size; (void)ws_size;
    const float* x        = (const float*)d_in[0];
    const int*   ei       = (const int*)  d_in[1];
    const float* enc_W    = (const float*)d_in[2];
    const float* enc_b    = (const float*)d_in[3];
    const float* gat_Wl   = (const float*)d_in[4];
    const float* gat_bl   = (const float*)d_in[5];
    const float* gat_Wr   = (const float*)d_in[6];
    const float* gat_br   = (const float*)d_in[7];
    const float* gat_att  = (const float*)d_in[8];
    const float* gat_bias = (const float*)d_in[9];
    const float* W1       = (const float*)d_in[10];
    const float* b1       = (const float*)d_in[11];
    const float* W2       = (const float*)d_in[12];
    const float* b2       = (const float*)d_in[13];
    const float* act_W    = (const float*)d_in[14];
    const float* act_b    = (const float*)d_in[15];
    const float* dec_W    = (const float*)d_in[16];
    const float* dec_b    = (const float*)d_in[17];
    float* out = (float*)d_out;

    // f32 workspace carve-out (~41.5 MB)
    float* ws = (float*)d_ws;
    size_t off = 0;
    float* h      = ws + off; off += (size_t)N_NODES * DIM;
    float* xl     = ws + off; off += (size_t)N_NODES * HD;
    float* xr     = ws + off; off += (size_t)N_NODES * HD;
    float* agg    = ws + off; off += (size_t)N_NODES * HD;
    float* hn1    = ws + off; off += (size_t)N_NODES * DIM;
    float* logitb = ws + off; off += (size_t)E_TOT * HEADS;
    float* mbuf   = ws + off; off += (size_t)N_NODES * HEADS;
    float* denom  = ws + off; off += (size_t)N_NODES * HEADS;
    float* total  = ws + off; off += (size_t)N_NODES;
    float* fin    = ws + off; off += (size_t)N_NODES * DIM;
    // f16 packed-weight carve-out (~116 KB), 32B-aligned (off is a multiple of 8)
    _Float16* hws = (_Float16*)(ws + off);
    size_t ho = 0;
    _Float16* p_enc = hws + ho; ho += (size_t)DIM * DIM;
    _Float16* p_wl  = hws + ho; ho += (size_t)DIM * HD;
    _Float16* p_wr  = hws + ho; ho += (size_t)DIM * HD;
    _Float16* p_w1  = hws + ho; ho += (size_t)HD  * DIM;
    _Float16* p_w2  = hws + ho; ho += (size_t)DIM * DIM;
    _Float16* p_dec = hws + ho; ho += (size_t)N_PRED * DIM * N_CLASS;

    const int TB = 128;                          // 4 waves / block for GEMMs
    auto gemm_grid = [](int M, int NC, int NT) {
        int waves = (M >> 4) * ((NC >> 4) / NT);
        return (waves + 3) / 4;
    };
    auto pack_grid = [](int K, int NC) {
        int tot = (NC >> 4) * (K >> 5) * 32;
        return (tot + 255) / 256;
    };

    // --- pack all weights into WMMA fragment order (once; weights are static)
    pack_b<<<pack_grid(DIM, DIM), 256, 0, stream>>>(enc_W, p_enc, DIM, DIM);
    pack_b<<<pack_grid(DIM, HD ), 256, 0, stream>>>(gat_Wl, p_wl, DIM, HD);
    pack_b<<<pack_grid(DIM, HD ), 256, 0, stream>>>(gat_Wr, p_wr, DIM, HD);
    pack_b<<<pack_grid(HD , DIM), 256, 0, stream>>>(W1, p_w1, HD, DIM);
    pack_b<<<pack_grid(DIM, DIM), 256, 0, stream>>>(W2, p_w2, DIM, DIM);
    pack_b<<<pack_grid(DIM, N_CLASS), 256, 0, stream>>>(dec_W, p_dec, DIM, N_CLASS);
    pack_b<<<pack_grid(DIM, N_CLASS), 256, 0, stream>>>(
        dec_W + (size_t)DIM * N_CLASS, p_dec + (size_t)DIM * N_CLASS, DIM, N_CLASS);

    // --- state init + encoder: h0 = x @ enc_W + enc_b
    init_state<<<(N_NODES * DIM + 255) / 256, 256, 0, stream>>>(total, fin);
    gemm_wmma<false, 4><<<gemm_grid(N_NODES, DIM, 4), TB, 0, stream>>>(
        x, p_enc, enc_b, nullptr, h, N_NODES, DIM, DIM);

    for (int step = 0; step < N_STEPS; ++step) {
        // xl = h @ Wl + bl ; xr = h @ Wr + br   ([N,256], L2-resident)
        gemm_wmma<false, 4><<<gemm_grid(N_NODES, HD, 4), TB, 0, stream>>>(
            h, p_wl, gat_bl, nullptr, xl, N_NODES, DIM, HD);
        gemm_wmma<false, 4><<<gemm_grid(N_NODES, HD, 4), TB, 0, stream>>>(
            h, p_wr, gat_br, nullptr, xr, N_NODES, DIM, HD);

        init_seg<<<(N_NODES * HD + 255) / 256, 256, 0, stream>>>(agg, mbuf, denom);

        edge_logits<<<((E_TOT * HEADS) + 255) / 256, 256, 0, stream>>>(
            ei, xl, xr, gat_att, logitb, mbuf);
        edge_exp<<<((E_TOT * HEADS) + 255) / 256, 256, 0, stream>>>(
            ei, logitb, mbuf, denom);
        edge_agg<<<((E_TOT * 32) + 255) / 256, 256, 0, stream>>>(
            ei, xl, logitb, denom, agg);

        // hn1 = relu((agg + gat_bias) @ W1 + b1)    (gat_bias folded into A-load)
        gemm_wmma<true, 4><<<gemm_grid(N_NODES, DIM, 4), TB, 0, stream>>>(
            agg, p_w1, b1, gat_bias, hn1, N_NODES, HD, DIM);
        // h = hn1 @ W2 + b2   (writes new node state in place of h)
        gemm_wmma<false, 4><<<gemm_grid(N_NODES, DIM, 4), TB, 0, stream>>>(
            hn1, p_w2, b2, nullptr, h, N_NODES, DIM, DIM);

        // ACT halting accumulation
        act_update<<<((N_NODES * 32) + 255) / 256, 256, 0, stream>>>(
            h, act_W, act_b, total, fin);
    }

    // fused decoder GEMM + log_softmax
    decoder_wmma<<<((N_PRED * (N_NODES >> 4)) + 3) / 4, TB, 0, stream>>>(
        fin, p_dec, dec_b, out);
}
